// BasisLinear_47510928228962
// MI455X (gfx1250) — matrix operand order repaired and checked by reading, not compiled, not run on hardware
//
#include <hip/hip_runtime.h>
#include <hip/hip_bf16.h>
#include <stdint.h>

// ---------------- problem constants (match reference) ----------------
#define N_ROWS   2048
#define IN_FEATS 1024
#define VOCAB    50000
#define NB       8
#define NC       512
#define FPB      128          // features per basis

typedef __attribute__((ext_vector_type(16))) _Float16 v16h;
typedef __attribute__((ext_vector_type(8)))  _Float16 v8h;
typedef __attribute__((ext_vector_type(4)))  _Float16 v4h;
typedef __attribute__((ext_vector_type(8)))  float    v8f;

// ---------------------------------------------------------------------
// Stage A: fp32 -> f16 conversion of x and weight (vectorized float4)
// ---------------------------------------------------------------------
__global__ void __launch_bounds__(256)
cvt_f16_kernel(const float* __restrict__ x, const float* __restrict__ w,
               _Float16* __restrict__ x16, _Float16* __restrict__ w16) {
    const int xq = (N_ROWS * IN_FEATS) / 4;   // 524288 float4s
    const int wq = (NB * NC * FPB) / 4;       // 131072 float4s
    int idx = blockIdx.x * blockDim.x + threadIdx.x;
    if (idx < xq) {
        float4 v = reinterpret_cast<const float4*>(x)[idx];
        v4h h = { (_Float16)v.x, (_Float16)v.y, (_Float16)v.z, (_Float16)v.w };
        reinterpret_cast<v4h*>(x16)[idx] = h;
    } else if (idx < xq + wq) {
        int j = idx - xq;
        float4 v = reinterpret_cast<const float4*>(w)[j];
        v4h h = { (_Float16)v.x, (_Float16)v.y, (_Float16)v.z, (_Float16)v.w };
        reinterpret_cast<v4h*>(w16)[j] = h;
    }
}

// ---------------------------------------------------------------------
// Stage B: per-basis GEMM with v_wmma_f32_16x16x32_f16.
// One wave -> 32 rows x 64 clusters (2 M-tiles x 4 N-tiles), K=128 in
// 4 steps of 32.  Epilogue adds bias, converts to f16 and stores into a
// row-chunk-tiled layout scores16[b][n>>3][c][n&7]: the 8 accumulator
// VGPRs of a lane are exactly one (b, n-chunk, c) group -> one
// global_store_b128 per tile, fully coalesced across lanes.
// ---------------------------------------------------------------------
__global__ void __launch_bounds__(128)
basis_gemm_wmma(const _Float16* __restrict__ x16,
                const _Float16* __restrict__ w16,
                const float*    __restrict__ bias,
                _Float16*       __restrict__ scores16) {
    const int lane   = threadIdx.x & 31;
    const int waveId = blockIdx.x * 4 + (threadIdx.x >> 5);   // 0..4095

    const int b     = waveId >> 9;          // 0..7
    const int rem   = waveId & 511;
    const int mBase = (rem >> 3) * 32;      // 64 M-groups of 32 rows
    const int nBase = (rem & 7) * 64;       // 8 N-groups of 64 clusters

    const int half = lane >> 4;             // 0/1 lane group
    const int lrow = lane & 15;

    v8f acc[2][4] = {};

    const _Float16* aRow0 = x16 + (size_t)(mBase + lrow) * IN_FEATS + b * FPB;
    const _Float16* aRow1 = aRow0 + (size_t)16 * IN_FEATS;
    const _Float16* bBase = w16 + ((size_t)(b * NC + nBase + lrow)) * FPB + half * 16;

    #pragma unroll
    for (int ks = 0; ks < 4; ++ks) {
        const int k0 = ks * 32;

        // A fragments: lane group h holds K = h*16 + 0..15 (contiguous)
        v16h a[2];
        #pragma unroll
        for (int i = 0; i < 2; ++i) {
            const _Float16* ap = (i == 0 ? aRow0 : aRow1) + k0 + half * 16;
            v8h lo = *(const v8h*)(ap);
            v8h hi = *(const v8h*)(ap + 8);
            #pragma unroll
            for (int t = 0; t < 8; ++t) { a[i][t] = lo[t]; a[i][t + 8] = hi[t]; }
        }

        // B fragments: lane group h holds K = h*16 + 0..15 (contiguous)
        v16h bb[4];
        #pragma unroll
        for (int j = 0; j < 4; ++j)
            bb[j] = *(const v16h*)(bBase + (size_t)j * 16 * FPB + k0);

        #pragma unroll
        for (int i = 0; i < 2; ++i)
            #pragma unroll
            for (int j = 0; j < 4; ++j)
                acc[i][j] = __builtin_amdgcn_wmma_f32_16x16x32_f16(
                    false, a[i], false, bb[j],
                    (short)0, acc[i][j], false, false);
    }

    float bv[4];
    #pragma unroll
    for (int j = 0; j < 4; ++j)
        bv[j] = bias[b * NC + nBase + j * 16 + lrow];

    // D layout: VGPR r holds M = r + half*8 -> n&7 == r, n>>3 == mBase/8 + i*2 + half
    #pragma unroll
    for (int i = 0; i < 2; ++i) {
        const int rowChunk = (mBase >> 3) + i * 2 + half;
        #pragma unroll
        for (int j = 0; j < 4; ++j) {
            const int c = nBase + j * 16 + lrow;
            v8h h;
            #pragma unroll
            for (int r = 0; r < 8; ++r)
                h[r] = (_Float16)(acc[i][j][r] + bv[j]);
            *(v8h*)(scores16 + (((size_t)(b * (N_ROWS / 8) + rowChunk)) * NC + c) * 8) = h;
        }
    }
}

// ---------------------------------------------------------------------
// Stage C: out[n][v] = sum_b scores[b][n][coords[b][v]]
// Block: NT=8 consecutive n-rows x VT=12500 vocab entries.
// Staging: scores16 is already tiled [b][nchunk][c][8 rows] f16, so the
// block's working set is 8 contiguous 8KB runs -> copied straight into
// 64KB LDS with global_load_async_to_lds_b128 (ASYNCcnt path).
// Gather: ONE ds_load_b128 per basis per v returns all 8 rows (v8h);
// 8-way basis sum done with packed-f16 adds, converted to f32 and
// streamed with non-temporal stores (keeps scores resident in L2).
// ---------------------------------------------------------------------
#define NT 8
#define VT 12500

__global__ void __launch_bounds__(256)
gather_sum_kernel(const _Float16* __restrict__ scores16,
                  const int*      __restrict__ coords,
                  float*          __restrict__ out) {
    __shared__ _Float16 smem[NB * NC * NT];   // 65536 bytes, [b][c][8 rows]

    const int tid    = threadIdx.x;
    const int nChunk = blockIdx.y;            // 0..255 (rows nChunk*8 .. +7)
    const int vBase  = blockIdx.x * VT;

    // ---- async global -> LDS staging: 4096 chunks of 16B, 16 per thread
    for (int chunk = tid; chunk < NB * NC; chunk += 256) {
        const int b   = chunk >> 9;           // basis
        const int c   = chunk & (NC - 1);     // cluster
        const _Float16* gp = scores16 +
            (((size_t)(b * (N_ROWS / 8) + nChunk)) * NC + c) * 8;
        uint32_t ldsAddr = (uint32_t)(uintptr_t)&smem[chunk * 8];
        uint64_t gAddr   = (uint64_t)(uintptr_t)gp;
        asm volatile("global_load_async_to_lds_b128 %0, %1, off"
                     :: "v"(ldsAddr), "v"(gAddr) : "memory");
    }
    asm volatile("s_wait_asynccnt 0" ::: "memory");
    __syncthreads();

    const v8h* lds8 = (const v8h*)smem;       // index: b*NC + c -> 8 rows
    int vEnd = vBase + VT;
    if (vEnd > VOCAB) vEnd = VOCAB;

    for (int v = vBase + tid; v < vEnd; v += 256) {
        v8h t0 = lds8[0 * NC + coords[0 * VOCAB + v]];
        v8h t1 = lds8[1 * NC + coords[1 * VOCAB + v]];
        v8h t2 = lds8[2 * NC + coords[2 * VOCAB + v]];
        v8h t3 = lds8[3 * NC + coords[3 * VOCAB + v]];
        v8h t4 = lds8[4 * NC + coords[4 * VOCAB + v]];
        v8h t5 = lds8[5 * NC + coords[5 * VOCAB + v]];
        v8h t6 = lds8[6 * NC + coords[6 * VOCAB + v]];
        v8h t7 = lds8[7 * NC + coords[7 * VOCAB + v]];

        // pairwise tree of packed-f16 adds (v_pk_add_f16)
        v8h s01 = t0 + t1, s23 = t2 + t3, s45 = t4 + t5, s67 = t6 + t7;
        v8h s   = (s01 + s23) + (s45 + s67);

        float* op = out + ((size_t)nChunk * NT) * VOCAB + v;
        #pragma unroll
        for (int r = 0; r < NT; ++r)
            __builtin_nontemporal_store((float)s[r], op + (size_t)r * VOCAB);
    }
}

// ---------------------------------------------------------------------
extern "C" void kernel_launch(void* const* d_in, const int* in_sizes, int n_in,
                              void* d_out, int out_size, void* d_ws, size_t ws_size,
                              hipStream_t stream) {
    const float* x      = (const float*)d_in[0];   // (2048, 1024) f32
    const float* w      = (const float*)d_in[1];   // (8, 512, 128) f32
    const float* bias   = (const float*)d_in[2];   // (8, 512) f32
    const int*   coords = (const int*)  d_in[3];   // (8, 50000) i32
    float*       out    = (float*)d_out;           // (2048, 50000) f32

    // workspace: x16 (4 MiB) | w16 (1 MiB) | scores16 tiled (16 MiB)
    char* ws = (char*)d_ws;
    _Float16* x16      = (_Float16*)ws;
    _Float16* w16      = (_Float16*)(ws + (size_t)4 * 1024 * 1024);
    _Float16* scores16 = (_Float16*)(ws + (size_t)5 * 1024 * 1024);

    // Stage A: convert (655360 float4 elements total)
    cvt_f16_kernel<<<2560, 256, 0, stream>>>(x, w, x16, w16);

    // Stage B: 4096 waves, 4 waves/block
    basis_gemm_wmma<<<1024, 128, 0, stream>>>(x16, w16, bias, scores16);

    // Stage C: grid (50000/VT, 2048/NT) = (4, 256)
    dim3 grid(VOCAB / VT, N_ROWS / NT);
    gather_sum_kernel<<<grid, 256, 0, stream>>>(scores16, coords, out);
}
